// NetworkGNN_82824149336546
// MI455X (gfx1250) — matrix operand order repaired and checked by reading, not compiled, not run on hardware
//
#include <hip/hip_runtime.h>
#include <hip/hip_bf16.h>

#define NN 50000
#define NE 600000
#define HID 128
#define NG 512
#define ODIM 128
#define BNEPS 1e-5f

typedef __attribute__((ext_vector_type(16))) __bf16 v16bf;
typedef __attribute__((ext_vector_type(8)))  float v8f;

static inline int nb(long n, int t) { return (int)((n + t - 1) / t); }

// ---------------- elementwise / utility kernels ----------------

__global__ void fill_kernel(float* p, float v, int n) {
    int i = blockIdx.x * blockDim.x + threadIdx.x;
    if (i < n) p[i] = v;
}

__global__ void copy_kernel(const float* __restrict__ s, float* __restrict__ d, int n) {
    int i = blockIdx.x * blockDim.x + threadIdx.x;
    if (i < n) d[i] = s[i];
}

__global__ void add_kernel(float* __restrict__ d, const float* __restrict__ s, int n) {
    int i = blockIdx.x * blockDim.x + threadIdx.x;
    if (i < n) d[i] += s[i];
}

__global__ void bcast_kernel(const float* __restrict__ v, float* __restrict__ out, int n, int C) {
    int i = blockIdx.x * blockDim.x + threadIdx.x;
    if (i < n) out[i] = v[i % C];
}

// APPNP combine: h = (1-alpha)*prop + alpha*h0
__global__ void appnp_kernel(const float* __restrict__ p, const float* __restrict__ h0,
                             float* __restrict__ h, long n) {
    long i = (long)blockIdx.x * blockDim.x + threadIdx.x;
    if (i < n) h[i] = 0.2f * p[i] + 0.8f * h0[i];
}

// ---------------- GCN normalization ----------------

__global__ void deg_kernel(const int* __restrict__ dst, float* __restrict__ deg, int E) {
    int e = blockIdx.x * blockDim.x + threadIdx.x;
    if (e < E) atomicAdd(&deg[dst[e]], 1.0f);
}

__global__ void dinv_kernel(const float* __restrict__ deg, float* __restrict__ dinv, int n) {
    int i = blockIdx.x * blockDim.x + threadIdx.x;
    if (i < n) dinv[i] = rsqrtf(fmaxf(deg[i], 1.0f));
}

__global__ void enorm_kernel(const int* __restrict__ src, const int* __restrict__ dst,
                             const float* __restrict__ dinv, float* __restrict__ en, int E) {
    int e = blockIdx.x * blockDim.x + threadIdx.x;
    if (e < E) en[e] = dinv[src[e]] * dinv[dst[e]];
}

// ---------------- propagation (A_hat @ h) ----------------

// self-loop term: out[i][:] = dinv[i]^2 * h[i][:]
__global__ void prop_self_kernel(const float* __restrict__ h, const float* __restrict__ dinv,
                                 float* __restrict__ out, long n) {
    long i = (long)blockIdx.x * blockDim.x + threadIdx.x;
    if (i < n) {
        float d = dinv[i >> 7];
        out[i] = d * d * h[i];
    }
}

// edge scatter-add: one wave per edge, 4 channels per lane (float4 gather)
__global__ void prop_edge_kernel(const int* __restrict__ src, const int* __restrict__ dst,
                                 const float* __restrict__ en, const float* __restrict__ h,
                                 float* __restrict__ out, int E) {
    long gid = (long)blockIdx.x * blockDim.x + threadIdx.x;
    int e = (int)(gid >> 5);
    if (e >= E) return;
    int lane = (int)(gid & 31);
    int s = src[e], d = dst[e];
    float nm = en[e];
    float4 v = ((const float4*)(h + (long)s * HID))[lane];
    float* o = out + (long)d * HID + lane * 4;
    atomicAdd(o + 0, nm * v.x);
    atomicAdd(o + 1, nm * v.y);
    atomicAdd(o + 2, nm * v.z);
    atomicAdd(o + 3, nm * v.w);
}

// ---------------- weight pre-pack into WMMA B-fragment order ----------------
// For W[K x N] (row-major f32) emit bf16 so that lane l of the wave owning
// column-tile ct / k-block kb reads its whole 16-element fragment contiguously:
//   P[(((kb*(N/16)) + ct)*32 + l)*16 + e] = W[(kb*32 + (e>>3)*16 + (l>>4)*8 + (e&7)) * N + ct*16 + (l&15)]

__global__ void packW_kernel(const float* __restrict__ Wf, __bf16* __restrict__ P, int K, int N) {
    int i = blockIdx.x * blockDim.x + threadIdx.x;
    if (i >= K * N) return;
    int e    = i & 15;
    int lane = (i >> 4) & 31;
    int rest = i >> 9;               // kb * (N/16) + ct
    int nt = N >> 4;
    int kb = rest / nt, ct = rest - kb * nt;
    int k = kb * 32 + (e >> 3) * 16 + (lane >> 4) * 8 + (e & 7);
    int n = ct * 16 + (lane & 15);
    P[i] = (__bf16)Wf[(long)k * N + n];
}

// ---------------- WMMA GEMM: C[M,N] = A[M,K] @ B[K,N] + bias[N] ----------------
// bf16 operands, f32 accumulation; A converted on the fly, B pre-packed.
// blockDim = (32,4): each wave owns a 16x16 output tile; block covers 16x64.
// K, N compile-time -> fully unrolled K-loop (4 or 8 v_wmma back-to-back).

template <int K, int N>
__global__ __launch_bounds__(128) void
wmma_gemm_bias(const float* __restrict__ A, const __bf16* __restrict__ Bp,
               const float* __restrict__ bias, float* __restrict__ C, int M) {
    constexpr int NT = N / 16;
    int rowT = blockIdx.x * 16;
    int ct   = blockIdx.y * 4 + threadIdx.y;   // column-tile index
    int lane = threadIdx.x;
    int r  = lane & 15;
    int kh = lane >> 4;

    v8f acc = {0.f, 0.f, 0.f, 0.f, 0.f, 0.f, 0.f, 0.f};
    const float* Arow = A + (long)(rowT + r) * K + kh * 8;
    #pragma unroll
    for (int kb = 0; kb < K / 32; ++kb) {
        v16bf a;
        #pragma unroll
        for (int e = 0; e < 8; ++e) a[e]     = (__bf16)Arow[kb * 32 + e];
        #pragma unroll
        for (int e = 0; e < 8; ++e) a[8 + e] = (__bf16)Arow[kb * 32 + 16 + e];
        v16bf bm = *(const v16bf*)(Bp + (((long)kb * NT + ct) * 32 + lane) * 16);
        acc = __builtin_amdgcn_wmma_f32_16x16x32_bf16(false, a, false, bm,
                                                      (short)0, acc, false, false);
    }
    int n = ct * 16 + r;
    float bv = bias[n];
    #pragma unroll
    for (int i = 0; i < 8; ++i)
        C[(long)(rowT + kh * 8 + i) * N + n] = acc[i] + bv;
}

// ---------------- BatchNorm (2-pass) ----------------

__global__ void colstat_kernel(const float* __restrict__ h, float* __restrict__ sum,
                               float* __restrict__ sumsq, int M, int C) {
    int c = threadIdx.x;
    float s = 0.f, s2 = 0.f;
    for (int r = blockIdx.x; r < M; r += gridDim.x) {
        float v = h[(long)r * C + c];
        s += v; s2 += v * v;
    }
    atomicAdd(&sum[c], s);
    atomicAdd(&sumsq[c], s2);
}

__global__ void bn_apply_kernel(float* __restrict__ h, const float* __restrict__ sum,
                                const float* __restrict__ sumsq,
                                const float* __restrict__ g, const float* __restrict__ bt,
                                const float* __restrict__ vn, const int* __restrict__ batch,
                                int M, int C, int relu) {
    long i = (long)blockIdx.x * blockDim.x + threadIdx.x;
    if (i >= (long)M * C) return;
    int c = (int)(i % C);
    int r = (int)(i / C);
    float inv = 1.0f / (float)M;
    float m  = sum[c] * inv;
    float var = sumsq[c] * inv - m * m;
    float rs = rsqrtf(var + BNEPS);
    float v = g[c] * (h[i] - m) * rs + bt[c];
    if (relu) v = fmaxf(v, 0.f);
    if (vn)   v += vn[(long)batch[r] * C + c];
    h[i] = v;
}

// ---------------- segment ops over graphs ----------------

__global__ void seg_add_kernel(const float* __restrict__ h, const int* __restrict__ batch,
                               float* __restrict__ out, long n) {
    long i = (long)blockIdx.x * blockDim.x + threadIdx.x;
    if (i < n) {
        int r = (int)(i >> 7), c = (int)(i & 127);
        atomicAdd(&out[(long)batch[r] * HID + c], h[i]);
    }
}

__global__ void count_kernel(const int* __restrict__ batch, float* __restrict__ cnt, int n) {
    int i = blockIdx.x * blockDim.x + threadIdx.x;
    if (i < n) atomicAdd(&cnt[batch[i]], 1.0f);
}

__global__ void pooldiv_kernel(float* __restrict__ pooled, const float* __restrict__ cnt, int n) {
    int i = blockIdx.x * blockDim.x + threadIdx.x;
    if (i < n) pooled[i] /= fmaxf(cnt[i >> 7], 1.0f);
}

// ---------------- host orchestration ----------------

extern "C" void kernel_launch(void* const* d_in, const int* in_sizes, int n_in,
                              void* d_out, int out_size, void* d_ws, size_t ws_size,
                              hipStream_t stream) {
    const float* x      = (const float*)d_in[0];
    const int*   ei     = (const int*)  d_in[1];
    const int*   batch  = (const int*)  d_in[2];
    const float* W      = (const float*)d_in[3];   // [3,128,128]
    const float* b      = (const float*)d_in[4];   // [3,128]
    const float* gamma  = (const float*)d_in[5];
    const float* beta   = (const float*)d_in[6];
    const float* vn_emb = (const float*)d_in[7];   // [128]
    const float* W1     = (const float*)d_in[8];   // [2,128,256]
    const float* b1     = (const float*)d_in[9];
    const float* g1     = (const float*)d_in[10];
    const float* bt1    = (const float*)d_in[11];
    const float* W2     = (const float*)d_in[12];  // [2,256,128]
    const float* b2     = (const float*)d_in[13];
    const float* g2     = (const float*)d_in[14];
    const float* bt2    = (const float*)d_in[15];
    const float* Wout   = (const float*)d_in[16];  // [128,128]
    const float* bout   = (const float*)d_in[17];
    float* out = (float*)d_out;

    const int* src = ei;
    const int* dst = ei + NE;

    // workspace carve-up (floats, then 64B-aligned bf16 packed-weight region)
    float* ws    = (float*)d_ws;
    float* deg   = ws;                   ws += NN;
    float* dinv  = ws;                   ws += NN;
    float* enorm = ws;                   ws += NE;
    float* hA    = ws;                   ws += (long)NN * HID;
    float* hB    = ws;                   ws += (long)NN * HID;
    float* h0    = ws;                   ws += (long)NN * HID;
    float* vn    = ws;                   ws += NG * HID;
    float* vt    = ws;                   ws += NG * HID;
    float* t1    = ws;                   ws += NG * 256;
    float* t2    = ws;                   ws += NG * HID;
    float* cnt   = ws;                   ws += NG;
    float* pooled= ws;                   ws += NG * HID;
    float* csum  = ws;                   ws += 256;
    float* csq   = ws;                   ws += 256;
    uintptr_t p = ((uintptr_t)ws + 63) & ~(uintptr_t)63;
    __bf16* pW    = (__bf16*)p;          // 3 * 128*128
    __bf16* pW1   = pW   + 3L * HID * HID;   // 2 * 128*256
    __bf16* pW2   = pW1  + 2L * HID * 256;   // 2 * 256*128
    __bf16* pWout = pW2  + 2L * 256 * HID;   // 128*128

    const long NH = (long)NN * HID;
    dim3 wb(32, 4);

    // --- pack all weights into WMMA fragment order (bf16) ---
    for (int i = 0; i < 3; ++i)
        packW_kernel<<<nb(HID*HID,256),256,0,stream>>>(W + (long)i*HID*HID,
                                                       pW + (long)i*HID*HID, HID, HID);
    for (int j = 0; j < 2; ++j) {
        packW_kernel<<<nb(HID*256,256),256,0,stream>>>(W1 + (long)j*HID*256,
                                                       pW1 + (long)j*HID*256, HID, 256);
        packW_kernel<<<nb(256*HID,256),256,0,stream>>>(W2 + (long)j*256*HID,
                                                       pW2 + (long)j*256*HID, 256, HID);
    }
    packW_kernel<<<nb(HID*ODIM,256),256,0,stream>>>(Wout, pWout, HID, ODIM);

    // --- GCN symmetric normalization (self-loops included) ---
    fill_kernel<<<nb(NN,256),256,0,stream>>>(deg, 1.0f, NN);  // self-loop degree
    deg_kernel<<<nb(NE,256),256,0,stream>>>(dst, deg, NE);
    dinv_kernel<<<nb(NN,256),256,0,stream>>>(deg, dinv, NN);
    enorm_kernel<<<nb(NE,256),256,0,stream>>>(src, dst, dinv, enorm, NE);

    // virtual node init
    bcast_kernel<<<nb(NG*HID,256),256,0,stream>>>(vn_emb, vn, NG*HID, HID);

    const float* hcur = x;
    for (int i = 0; i < 2; ++i) {
        // prop(h) -> hB
        prop_self_kernel<<<nb(NH,256),256,0,stream>>>(hcur, dinv, hB, NH);
        prop_edge_kernel<<<nb((long)NE*32,256),256,0,stream>>>(src, dst, enorm, hcur, hB, NE);
        // hA = hB @ W[i] + b[i]
        wmma_gemm_bias<HID,HID><<<dim3(NN/16, HID/64), wb, 0, stream>>>(
            hB, pW + (long)i*HID*HID, b + i*HID, hA, NN);
        // BN + relu + vn[batch]
        fill_kernel<<<1,256,0,stream>>>(csum, 0.f, HID);
        fill_kernel<<<1,256,0,stream>>>(csq,  0.f, HID);
        colstat_kernel<<<256, HID, 0, stream>>>(hA, csum, csq, NN, HID);
        bn_apply_kernel<<<nb(NH,256),256,0,stream>>>(hA, csum, csq,
            gamma + i*HID, beta + i*HID, vn, batch, NN, HID, 1);
        // vt = vn + segment_sum(hA, batch)
        copy_kernel<<<nb(NG*HID,256),256,0,stream>>>(vn, vt, NG*HID);
        seg_add_kernel<<<nb(NH,256),256,0,stream>>>(hA, batch, vt, NH);
        // virtual-node MLP (torch negative-index wraparound: i=0 -> j=1, i=1 -> j=0)
        int j = (i == 0) ? 1 : 0;
        wmma_gemm_bias<HID,256><<<dim3(NG/16, 256/64), wb, 0, stream>>>(
            vt, pW1 + (long)j*HID*256, b1 + j*256, t1, NG);
        fill_kernel<<<1,256,0,stream>>>(csum, 0.f, 256);
        fill_kernel<<<1,256,0,stream>>>(csq,  0.f, 256);
        colstat_kernel<<<64, 256, 0, stream>>>(t1, csum, csq, NG, 256);
        bn_apply_kernel<<<nb(NG*256,256),256,0,stream>>>(t1, csum, csq,
            g1 + j*256, bt1 + j*256, nullptr, nullptr, NG, 256, 1);
        wmma_gemm_bias<256,HID><<<dim3(NG/16, HID/64), wb, 0, stream>>>(
            t1, pW2 + (long)j*256*HID, b2 + j*HID, t2, NG);
        fill_kernel<<<1,256,0,stream>>>(csum, 0.f, HID);
        fill_kernel<<<1,256,0,stream>>>(csq,  0.f, HID);
        colstat_kernel<<<64, HID, 0, stream>>>(t2, csum, csq, NG, HID);
        bn_apply_kernel<<<nb(NG*HID,256),256,0,stream>>>(t2, csum, csq,
            g2 + j*HID, bt2 + j*HID, nullptr, nullptr, NG, HID, 1);
        add_kernel<<<nb(NG*HID,256),256,0,stream>>>(vn, t2, NG*HID);
        hcur = hA;
    }

    // last conv: h0 = BN(prop(h) @ W[2] + b[2])   (no relu, no vn)
    prop_self_kernel<<<nb(NH,256),256,0,stream>>>(hcur, dinv, hB, NH);
    prop_edge_kernel<<<nb((long)NE*32,256),256,0,stream>>>(src, dst, enorm, hcur, hB, NE);
    wmma_gemm_bias<HID,HID><<<dim3(NN/16, HID/64), wb, 0, stream>>>(
        hB, pW + 2L*HID*HID, b + 2*HID, h0, NN);
    fill_kernel<<<1,256,0,stream>>>(csum, 0.f, HID);
    fill_kernel<<<1,256,0,stream>>>(csq,  0.f, HID);
    colstat_kernel<<<256, HID, 0, stream>>>(h0, csum, csq, NN, HID);
    bn_apply_kernel<<<nb(NH,256),256,0,stream>>>(h0, csum, csq,
        gamma + 2*HID, beta + 2*HID, nullptr, nullptr, NN, HID, 0);

    // APPNP: h_{k+1} = 0.2 * A_hat h_k + 0.8 * h0
    copy_kernel<<<nb(NH,256),256,0,stream>>>(h0, hA, NH);
    for (int k = 0; k < 5; ++k) {
        prop_self_kernel<<<nb(NH,256),256,0,stream>>>(hA, dinv, hB, NH);
        prop_edge_kernel<<<nb((long)NE*32,256),256,0,stream>>>(src, dst, enorm, hA, hB, NE);
        appnp_kernel<<<nb(NH,256),256,0,stream>>>(hB, h0, hA, NH);
    }

    // global mean pool + linear head
    fill_kernel<<<nb(NG,256),256,0,stream>>>(cnt, 0.f, NG);
    count_kernel<<<nb(NN,256),256,0,stream>>>(batch, cnt, NN);
    fill_kernel<<<nb(NG*HID,256),256,0,stream>>>(pooled, 0.f, NG*HID);
    seg_add_kernel<<<nb(NH,256),256,0,stream>>>(hA, batch, pooled, NH);
    pooldiv_kernel<<<nb(NG*HID,256),256,0,stream>>>(pooled, cnt, NG*HID);
    wmma_gemm_bias<HID,ODIM><<<dim3(NG/16, ODIM/64), wb, 0, stream>>>(
        pooled, pWout, bout, out, NG);
}
